// GraphConvLayer_71047349010846
// MI455X (gfx1250) — compile-verified, matrix operand order checked by reading, not run on hardware
//
#include <hip/hip_runtime.h>
#include <hip/hip_bf16.h>

// Problem constants (match reference)
#define BB 2
#define NN 20000
#define EE 256000
#define DD 256
#define HH 256
#define K2 512   // D + H

// LDS row strides (bytes) with 16B pad per row for bank-conflict-free ds_load_b128
#define W1_LDS_STRIDE 528    // 256*2 + 16
#define W2_LDS_STRIDE 1040   // 512*2 + 16

typedef __attribute__((ext_vector_type(16))) __bf16 v16bf;
typedef __attribute__((ext_vector_type(8)))  __bf16 v8bf;
typedef __attribute__((ext_vector_type(8)))  float  v8f;

typedef __attribute__((ext_vector_type(4))) unsigned int u32x4;
typedef __attribute__((ext_vector_type(8))) int          i32x8;
typedef __attribute__((ext_vector_type(4))) int          i32x4;

// ---------------- TDM: DMA a [rows x elemsPerRow] bf16 matrix (row-major, dense in
// global) into LDS offset 0, inserting `padAmount+1` dwords of LDS padding after
// every 2^(padInterval+1) dwords (== one row). Issued by one wave; tracked by TENSORcnt.
__device__ __forceinline__ void tdm_load_weights(const void* gptr,
                                                 unsigned elemsPerRow, unsigned rows,
                                                 unsigned padInterval, unsigned padAmount) {
    unsigned long long ga = (unsigned long long)gptr;
    u32x4 g0;
    g0[0] = 1u;                                           // count=1 (valid), user mode
    g0[1] = 0u;                                           // lds_addr = 0 (dyn-LDS base)
    g0[2] = (unsigned)(ga & 0xFFFFFFFFu);                 // global_addr[31:0]
    g0[3] = (unsigned)((ga >> 32) & 0x01FFFFFFu)          // global_addr[56:32]
          | (2u << 30);                                   // type = 2 ("image")
    i32x8 g1;
    g1[0] = (int)((1u << 16)                              // data_size = 2 bytes
          | (1u << 20)                                    // pad_enable
          | (padInterval << 22) | (padAmount << 25));
    g1[1] = (int)((elemsPerRow & 0xFFFFu) << 16);         // tensor_dim0[15:0]
    g1[2] = (int)(((elemsPerRow >> 16) & 0xFFFFu)         // tensor_dim0[31:16]
          | ((rows & 0xFFFFu) << 16));                    // tensor_dim1[15:0]
    g1[3] = (int)(((rows >> 16) & 0xFFFFu)                // tensor_dim1[31:16]
          | ((elemsPerRow & 0xFFFFu) << 16));             // tile_dim0
    g1[4] = (int)(rows & 0xFFFFu);                        // tile_dim1 (tile_dim2=0)
    g1[5] = (int)elemsPerRow;                             // tensor_dim0_stride[31:0]
    g1[6] = 0;
    g1[7] = 0;
    i32x4 z4 = {0, 0, 0, 0};
#if defined(__clang_major__) && (__clang_major__ >= 23)
    i32x8 z8 = {0, 0, 0, 0, 0, 0, 0, 0};
    __builtin_amdgcn_tensor_load_to_lds(g0, g1, z4, z4, z8, 0);
#else
    __builtin_amdgcn_tensor_load_to_lds(g0, g1, z4, z4, 0);
#endif
    __builtin_amdgcn_s_wait_tensorcnt(0);
}

// ---------------- prep kernels ----------------

__global__ void prep_affine(const float* __restrict__ g1, const float* __restrict__ be1,
                            const float* __restrict__ m1, const float* __restrict__ va1,
                            const float* __restrict__ g2, const float* __restrict__ be2,
                            const float* __restrict__ m2, const float* __restrict__ va2,
                            float* __restrict__ scale1, float* __restrict__ shift1,
                            float* __restrict__ scale2, float* __restrict__ shift2) {
    int i = threadIdx.x;
    if (i < DD) {
        float s = g1[i] * rsqrtf(va1[i] + 1e-3f);
        scale1[i] = s;
        shift1[i] = be1[i] - m1[i] * s;
    }
    if (i < K2) {
        float s = g2[i] * rsqrtf(va2[i] + 1e-3f);
        scale2[i] = s;
        shift2[i] = be2[i] - m2[i] * s;
    }
}

// W1 [D,H] f32 -> W1T [H,D] bf16
__global__ void transpose_w1(const float* __restrict__ W1, __bf16* __restrict__ W1T) {
    int i = blockIdx.x * blockDim.x + threadIdx.x;   // over D*H
    int d = i / HH, h = i % HH;
    W1T[(size_t)h * DD + d] = (__bf16)W1[i];
}

// W2 [K2,H] f32 -> W2T [H,K2] bf16
__global__ void transpose_w2(const float* __restrict__ W2, __bf16* __restrict__ W2T) {
    int i = blockIdx.x * blockDim.x + threadIdx.x;   // over K2*H
    int k = i / HH, h = i % HH;
    W2T[(size_t)h * K2 + k] = (__bf16)W2[i];
}

// per-target-node edge counts (float so the mean divide is a mul)
__global__ void count_edges(const int* __restrict__ tgt, float* __restrict__ cnt) {
    int i = blockIdx.x * blockDim.x + threadIdx.x;   // over B*E
    if (i < BB * EE) {
        int b = i / EE;
        atomicAdd(cnt + (size_t)b * NN + tgt[i], 1.0f);
    }
}

// ---------------- GEMM1: gather -> BN -> bf16 WMMA (B from LDS) -> relu -> *w -> atomic scatter
// block = 256 threads = 8 waves; tile = 64 edges x 256 cols; wave = 16 edges x 128 cols.
// W1T (128KB + row pad) is TDM-DMA'd into LDS once per workgroup.

__global__ __launch_bounds__(256) void gemm1_msg(
    const float* __restrict__ node_reps,       // [B,N,D]
    const int*   __restrict__ nbr_idx,         // [B,E]
    const int*   __restrict__ tgt_idx,         // [B,E]
    const float* __restrict__ ew,              // [B,E]
    const float* __restrict__ scale1, const float* __restrict__ shift1,  // [D]
    const __bf16* __restrict__ W1T,            // [H,D] bf16 (global)
    const float* __restrict__ b1,              // [H]
    float* __restrict__ sum)                   // [B,N,H] accumulator
{
    extern __shared__ char smem[];             // H rows x W1_LDS_STRIDE bytes

    const int b       = blockIdx.y;
    const int e0      = blockIdx.x * 64;
    const int wave    = threadIdx.x >> 5;
    const int lane    = threadIdx.x & 31;
    const int waveRow = wave & 3;              // 0..3 -> 16-edge sub-tile
    const int waveCol = wave >> 2;             // 0..1 -> 128-col half
    const int laneM   = lane & 15;
    const int hi      = lane >> 4;             // K-half select per ISA striping

    // one wave DMAs the whole weight matrix into LDS (pad 4 dwords per 128-dword row)
    if (wave == 0) {
        tdm_load_weights(W1T, DD, HH, /*padInterval 128dw*/6u, /*padAmount 4dw*/3u);
    }
    __syncthreads();

    // gather pointer for this lane's A row (row = laneM within the 16-edge tile)
    const int eA   = e0 + waveRow * 16 + laneM;
    const int nidx = nbr_idx[(size_t)b * EE + eA];
    const float* __restrict__ arow = node_reps + ((size_t)b * NN + nidx) * DD;

    v8f acc[8] = {};

    #pragma unroll
    for (int ks = 0; ks < 8; ++ks) {
        const int k0 = ks * 32;
        const int ka = k0 + hi * 8;            // lanes<16: {0..7,16..23}; lanes>=16: {8..15,24..31}
        const int kb = ka + 16;

        v8f xlo = *(const v8f*)(arow + ka);
        v8f xhi = *(const v8f*)(arow + kb);
        v8f slo = *(const v8f*)(scale1 + ka);
        v8f shl = *(const v8f*)(shift1 + ka);
        v8f shi = *(const v8f*)(scale1 + kb);
        v8f shh = *(const v8f*)(shift1 + kb);
        xlo = xlo * slo + shl;
        xhi = xhi * shi + shh;

        v16bf afrag;
        #pragma unroll
        for (int i = 0; i < 8; ++i) {
            afrag[i]     = (__bf16)xlo[i];
            afrag[i + 8] = (__bf16)xhi[i];
        }

        #pragma unroll
        for (int nt = 0; nt < 8; ++nt) {
            const char* rb = smem + (size_t)(waveCol * 128 + nt * 16 + laneM) * W1_LDS_STRIDE
                                  + 2 * ka;
            v8bf blo = *(const v8bf*)(rb);
            v8bf bhi = *(const v8bf*)(rb + 32);
            v16bf bfrag;
            #pragma unroll
            for (int i = 0; i < 8; ++i) {
                bfrag[i]     = blo[i];
                bfrag[i + 8] = bhi[i];
            }
            acc[nt] = __builtin_amdgcn_wmma_f32_16x16x32_bf16(
                false, afrag, false, bfrag, (short)0, acc[nt], false, false);
        }
    }

    // epilogue: +bias, relu, * edge weight, atomic scatter-add into sum[b, tgt, :]
    // C layout: vgpr j of acc -> row (j + 8*hi), col = waveCol*128 + nt*16 + laneM
    int   tid8[8];
    float w8[8];
    const int eC = e0 + waveRow * 16 + hi * 8;
    #pragma unroll
    for (int j = 0; j < 8; ++j) {
        tid8[j] = tgt_idx[(size_t)b * EE + eC + j];
        w8[j]   = ew[(size_t)b * EE + eC + j];
    }
    #pragma unroll
    for (int nt = 0; nt < 8; ++nt) {
        const int col  = waveCol * 128 + nt * 16 + laneM;
        const float bias = b1[col];
        #pragma unroll
        for (int j = 0; j < 8; ++j) {
            float v = acc[nt][j] + bias;
            v = v > 0.0f ? v : 0.0f;
            v *= w8[j];
            atomicAdd(sum + ((size_t)b * NN + tid8[j]) * HH + col, v);
        }
    }
}

// ---------------- GEMM2: concat(node_reps, sum/cnt) -> BN -> bf16 WMMA (B from LDS) -> relu
// tile = 64 rows x 256 cols per block, K = 512. W2T (256KB + pad) TDM-DMA'd into LDS.

__global__ __launch_bounds__(256) void gemm2_update(
    const float* __restrict__ node_reps,       // [B,N,D]
    const float* __restrict__ sum,             // [B,N,H]
    const float* __restrict__ cnt,             // [B,N]
    const float* __restrict__ scale2, const float* __restrict__ shift2,  // [K2]
    const __bf16* __restrict__ W2T,            // [H,K2] bf16 (global)
    const float* __restrict__ b2,              // [H]
    float* __restrict__ out)                   // [B,N,H]
{
    extern __shared__ char smem[];             // H rows x W2_LDS_STRIDE bytes

    const int b       = blockIdx.y;
    const int r0      = blockIdx.x * 64;
    const int wave    = threadIdx.x >> 5;
    const int lane    = threadIdx.x & 31;
    const int waveRow = wave & 3;
    const int waveCol = wave >> 2;
    const int laneM   = lane & 15;
    const int hi      = lane >> 4;

    if (wave == 0) {
        tdm_load_weights(W2T, K2, HH, /*padInterval 256dw*/7u, /*padAmount 4dw*/3u);
    }
    __syncthreads();

    int rowA = r0 + waveRow * 16 + laneM;
    if (rowA > NN - 1) rowA = NN - 1;          // clamp loads for partial tail tile
    const float* __restrict__ nrow = node_reps + ((size_t)b * NN + rowA) * DD;
    const float* __restrict__ srow = sum       + ((size_t)b * NN + rowA) * HH;
    const float c   = cnt[(size_t)b * NN + rowA];
    const float inv = 1.0f / (c > 1.0f ? c : 1.0f);

    v8f acc[8] = {};

    #pragma unroll
    for (int ks = 0; ks < 16; ++ks) {
        const int k0 = ks * 32;
        const int ka = k0 + hi * 8;
        const int kb = ka + 16;

        v8f xlo = (ka < DD) ? *(const v8f*)(nrow + ka)
                            : (*(const v8f*)(srow + (ka - DD))) * inv;
        v8f xhi = (kb < DD) ? *(const v8f*)(nrow + kb)
                            : (*(const v8f*)(srow + (kb - DD))) * inv;
        v8f slo = *(const v8f*)(scale2 + ka);
        v8f shl = *(const v8f*)(shift2 + ka);
        v8f shi = *(const v8f*)(scale2 + kb);
        v8f shh = *(const v8f*)(shift2 + kb);
        xlo = xlo * slo + shl;
        xhi = xhi * shi + shh;

        v16bf afrag;
        #pragma unroll
        for (int i = 0; i < 8; ++i) {
            afrag[i]     = (__bf16)xlo[i];
            afrag[i + 8] = (__bf16)xhi[i];
        }

        #pragma unroll
        for (int nt = 0; nt < 8; ++nt) {
            const char* rb = smem + (size_t)(waveCol * 128 + nt * 16 + laneM) * W2_LDS_STRIDE
                                  + 2 * ka;
            v8bf blo = *(const v8bf*)(rb);
            v8bf bhi = *(const v8bf*)(rb + 32);
            v16bf bfrag;
            #pragma unroll
            for (int i = 0; i < 8; ++i) {
                bfrag[i]     = blo[i];
                bfrag[i + 8] = bhi[i];
            }
            acc[nt] = __builtin_amdgcn_wmma_f32_16x16x32_bf16(
                false, afrag, false, bfrag, (short)0, acc[nt], false, false);
        }
    }

    // epilogue: +bias, relu, store (guard tail rows)
    #pragma unroll
    for (int nt = 0; nt < 8; ++nt) {
        const int col   = waveCol * 128 + nt * 16 + laneM;
        const float bias = b2[col];
        #pragma unroll
        for (int j = 0; j < 8; ++j) {
            const int rowC = r0 + waveRow * 16 + hi * 8 + j;
            if (rowC < NN) {
                float v = acc[nt][j] + bias;
                v = v > 0.0f ? v : 0.0f;
                out[((size_t)b * NN + rowC) * HH + col] = v;
            }
        }
    }
}

// ---------------- launcher ----------------

extern "C" void kernel_launch(void* const* d_in, const int* in_sizes, int n_in,
                              void* d_out, int out_size, void* d_ws, size_t ws_size,
                              hipStream_t stream) {
    const float* node_reps   = (const float*)d_in[0];
    const int*   node_idx    = (const int*)  d_in[1];
    const int*   nbr_idx     = (const int*)  d_in[2];
    const float* ew          = (const float*)d_in[3];
    const float* bn1_gamma   = (const float*)d_in[4];
    const float* bn1_beta    = (const float*)d_in[5];
    const float* bn1_mean    = (const float*)d_in[6];
    const float* bn1_var     = (const float*)d_in[7];
    const float* W1          = (const float*)d_in[8];
    const float* b1          = (const float*)d_in[9];
    const float* bn2_gamma   = (const float*)d_in[10];
    const float* bn2_beta    = (const float*)d_in[11];
    const float* bn2_mean    = (const float*)d_in[12];
    const float* bn2_var     = (const float*)d_in[13];
    const float* W2          = (const float*)d_in[14];
    const float* b2          = (const float*)d_in[15];
    float* out = (float*)d_out;

    // workspace layout (all offsets 256B aligned)
    char* ws = (char*)d_ws;
    const size_t sumBytes   = (size_t)BB * NN * HH * sizeof(float);   // 40,960,000
    const size_t cntBytes   = (size_t)BB * NN * sizeof(float);        //    160,000
    float*  sum    = (float*)(ws);
    float*  cnt    = (float*)(ws + sumBytes);
    float*  scale1 = (float*)(ws + sumBytes + cntBytes);
    float*  shift1 = scale1 + 256;
    float*  scale2 = shift1 + 256;
    float*  shift2 = scale2 + 512;
    __bf16* W1T    = (__bf16*)(shift2 + 512);                         // [H,D]
    __bf16* W2T    = W1T + (size_t)HH * DD;                           // [H,K2]

    // zero the scatter accumulators (sum + count are contiguous)
    hipMemsetAsync(sum, 0, sumBytes + cntBytes, stream);

    prep_affine<<<1, 512, 0, stream>>>(bn1_gamma, bn1_beta, bn1_mean, bn1_var,
                                       bn2_gamma, bn2_beta, bn2_mean, bn2_var,
                                       scale1, shift1, scale2, shift2);
    transpose_w1<<<(DD * HH) / 256, 256, 0, stream>>>(W1, W1T);
    transpose_w2<<<(K2 * HH) / 256, 256, 0, stream>>>(W2, W2T);
    count_edges<<<(BB * EE + 255) / 256, 256, 0, stream>>>(node_idx, cnt);

    gemm1_msg<<<dim3(EE / 64, BB), 256, (size_t)HH * W1_LDS_STRIDE, stream>>>(
        node_reps, nbr_idx, node_idx, ew, scale1, shift1, W1T, b1, sum);

    gemm2_update<<<dim3((NN + 63) / 64, BB), 256, (size_t)HH * W2_LDS_STRIDE, stream>>>(
        node_reps, sum, cnt, scale2, shift2, W2T, b2, out);
}
